// LaplacianPositionalTransformer_21861383537289
// MI455X (gfx1250) — compile-verified, matrix operand order checked by reading, not run on hardware
//
#include <hip/hip_runtime.h>
#include <hip/hip_bf16.h>

#define HD128 128
#define NNODES 50000
#define NEDGES 800000
#define SEQ 100

typedef __attribute__((ext_vector_type(16))) _Float16 v16h;
typedef __attribute__((ext_vector_type(8)))  float    v8f;

// ---------------- WMMA fragment helpers (gfx1250, wave32) ----------------
// A fragment 16x32 (MxK), f16, per ISA 7.12.2:
//  lanes 0-15: row = lane, K = {0..7} in v0..v3 halves, {16..23} in v4..v7
//  lanes 16-31: row = lane-16, K = {8..15} and {24..31}
__device__ __forceinline__ v16h load_frag(const float* __restrict__ src, int ld,
                                          int lane, int k0, int kmax) {
  int row = lane & 15;
  int kb  = (lane < 16) ? 0 : 8;
  v16h a;
#pragma unroll
  for (int j = 0; j < 16; ++j) {
    int k = k0 + kb + (j & 7) + ((j >> 3) << 4);
    a[j] = (_Float16)((k < kmax) ? src[(long)row * ld + k] : 0.f);
  }
  return a;
}

// D/C fragment 16x16 f32: lane holds col n = lane&15; VGPR r holds row r (lanes0-15) / r+8 (lanes16-31)
__device__ __forceinline__ void store_frag(float* __restrict__ dst, int ld, int col0,
                                           v8f d, int lane) {
  int n = col0 + (lane & 15);
  int rb = (lane < 16) ? 0 : 8;
#pragma unroll
  for (int r = 0; r < 8; ++r) dst[(long)(rb + r) * ld + n] = d[r];
}

__device__ __forceinline__ v8f splat8(float v) {
  v8f r = {v, v, v, v, v, v, v, v};
  return r;
}

__device__ __forceinline__ v8f wmma32(v16h a, v16h b, v8f c) {
  // D = A(16x32) * B(32x16) + C  -> v_wmma_f32_16x16x32_f16
  return __builtin_amdgcn_wmma_f32_16x16x32_f16(false, a, false, b, (short)0, c,
                                                false, false);
}

__device__ __forceinline__ void atomicMaxF(float* addr, float v) {
  if (v >= 0.f) atomicMax((int*)addr, __float_as_int(v));
  else          atomicMin((unsigned int*)addr, (unsigned int)__float_as_int(v));
}

// ---------------- K0: workspace init ----------------
__global__ __launch_bounds__(256) void k_init(float* loop_sum, float* outg, float* deg,
                                              float* denom, float* mbuf, float* bnsum,
                                              float* bnsq) {
  long stride = (long)gridDim.x * blockDim.x;
  for (long i = (long)blockIdx.x * blockDim.x + threadIdx.x; i < (long)NNODES * HD128;
       i += stride) {
    loop_sum[i] = 0.f;
    outg[i] = 0.f;
    if (i < NNODES) { deg[i] = 0.f; denom[i] = 0.f; mbuf[i] = -1e30f; }
    if (i < HD128) { bnsum[i] = 0.f; bnsq[i] = 0.f; }
  }
}

// ---------------- K1: fused node pipeline ----------------
// t = [node_enc(x) ; pos_enc(lap)] ; nf = comb(t) ; h = nf @ gat_lin_w^T
// a_src = h@att_src ; a_dst = h@att_dst.  One wave per 16-node tile, WMMA end-to-end.
__global__ __launch_bounds__(128) void k_nodes(
    const float* __restrict__ x, const float* __restrict__ lap,
    const float* __restrict__ Wn, const float* __restrict__ bn,
    const float* __restrict__ Wp, const float* __restrict__ bp,
    const float* __restrict__ Wc, const float* __restrict__ bc,
    const float* __restrict__ Wg, const float* __restrict__ att_src,
    const float* __restrict__ att_dst, float* __restrict__ hout,
    float* __restrict__ asrc, float* __restrict__ adst) {
  __shared__ float stage[4][16 * HD128];
  int wid = threadIdx.x >> 5, lane = threadIdx.x & 31;
  int tile = blockIdx.x * 4 + wid;
  if (tile * 16 >= NNODES) return;
  int row0 = tile * 16;
  float* sg = stage[wid];

  // --- node_enc: x[16x64] @ Wn^T(64x64) + bn -> cols 0..63 of t
  v16h ax0 = load_frag(x + (long)row0 * 64, 64, lane, 0, 64);
  v16h ax1 = load_frag(x + (long)row0 * 64, 64, lane, 32, 64);
#pragma unroll
  for (int j = 0; j < 4; ++j) {
    v8f acc = splat8(bn[j * 16 + (lane & 15)]);
    v16h b0 = load_frag(Wn + (long)(j * 16) * 64, 64, lane, 0, 64);
    v16h b1 = load_frag(Wn + (long)(j * 16) * 64, 64, lane, 32, 64);
    acc = wmma32(ax0, b0, acc);
    acc = wmma32(ax1, b1, acc);
    store_frag(sg, HD128, j * 16, acc, lane);
  }
  // --- pos_enc: lap[16x8] @ Wp^T(64x8) + bp -> cols 64..127 of t (K padded to 32)
  v16h ap = load_frag(lap + (long)row0 * 8, 8, lane, 0, 8);
#pragma unroll
  for (int j = 0; j < 4; ++j) {
    v8f acc = splat8(bp[j * 16 + (lane & 15)]);
    v16h b0 = load_frag(Wp + (long)(j * 16) * 8, 8, lane, 0, 8);
    acc = wmma32(ap, b0, acc);
    store_frag(sg, HD128, 64 + j * 16, acc, lane);
  }
  // --- comb: nf = t @ Wc^T + bc  (K=128 -> 4 WMMA k-steps)
  v16h at[4];
#pragma unroll
  for (int ks = 0; ks < 4; ++ks) at[ks] = load_frag(sg, HD128, lane, ks * 32, HD128);
  v8f nf2[8];
#pragma unroll
  for (int j = 0; j < 8; ++j) {
    v8f acc = splat8(bc[j * 16 + (lane & 15)]);
#pragma unroll
    for (int ks = 0; ks < 4; ++ks) {
      v16h b = load_frag(Wc + (long)(j * 16) * HD128, HD128, lane, ks * 32, HD128);
      acc = wmma32(at[ks], b, acc);
    }
    nf2[j] = acc;
  }
#pragma unroll
  for (int j = 0; j < 8; ++j) store_frag(sg, HD128, j * 16, nf2[j], lane);
  // --- h = nf @ Wg^T (no bias)
#pragma unroll
  for (int ks = 0; ks < 4; ++ks) at[ks] = load_frag(sg, HD128, lane, ks * 32, HD128);
#pragma unroll
  for (int j = 0; j < 8; ++j) {
    v8f acc = splat8(0.f);
#pragma unroll
    for (int ks = 0; ks < 4; ++ks) {
      v16h b = load_frag(Wg + (long)(j * 16) * HD128, HD128, lane, ks * 32, HD128);
      acc = wmma32(at[ks], b, acc);
    }
    store_frag(hout + (long)row0 * HD128, HD128, j * 16, acc, lane);
    store_frag(sg, HD128, j * 16, acc, lane);
  }
  // --- attention pre-dots (half-wave does src, half does dst)
  if (lane < 16) {
    float s = 0.f;
    for (int c = 0; c < HD128; ++c) s += sg[lane * HD128 + c] * att_src[c];
    asrc[row0 + lane] = s;
  } else {
    int r = lane - 16;
    float s = 0.f;
    for (int c = 0; c < HD128; ++c) s += sg[r * HD128 + c] * att_dst[c];
    adst[row0 + r] = s;
  }
}

// ---------------- w_eff = gat_lin_edge_w^T @ att_edge (folds ee away) ----------------
__global__ void k_weff(const float* __restrict__ We, const float* __restrict__ att_edge,
                       float* __restrict__ w_eff) {
  int k = threadIdx.x;
  float s = 0.f;
  for (int n = 0; n < HD128; ++n) s += att_edge[n] * We[(long)n * HD128 + k];
  w_eff[k] = s;
}

// ---------------- K2: fused edge pipeline ----------------
// ea = edge_enc(edge_attr) tile-resident (never stored); scatter to loop_sum/deg;
// a_edge = ea . w_eff; alpha = leaky(asrc[src]+adst[dst]+a_edge); atomic-max m[dst].
__global__ __launch_bounds__(128) void k_edges(
    const float* __restrict__ eattr, const int* __restrict__ eidx,
    const float* __restrict__ We, const float* __restrict__ be,
    const float* __restrict__ w_eff, const float* __restrict__ asrc,
    const float* __restrict__ adst, float* __restrict__ loop_sum,
    float* __restrict__ deg, float* __restrict__ alpha, float* __restrict__ mbuf) {
  __shared__ float stage[4][16 * HD128];
  int wid = threadIdx.x >> 5, lane = threadIdx.x & 31;
  int tile = blockIdx.x * 4 + wid;
  if (tile * 16 >= NEDGES) return;
  int e0 = tile * 16;
  float* sg = stage[wid];

  v16h ae = load_frag(eattr + (long)e0 * 16, 16, lane, 0, 16);  // K=16 padded to 32
#pragma unroll
  for (int j = 0; j < 8; ++j) {
    v8f acc = splat8(be[j * 16 + (lane & 15)]);
    v16h b = load_frag(We + (long)(j * 16) * 16, 16, lane, 0, 16);
    acc = wmma32(ae, b, acc);
    store_frag(sg, HD128, j * 16, acc, lane);
  }
  // scatter ea into loop_sum[dst] (lane owns 4 channels)
#pragma unroll 4
  for (int r = 0; r < 16; ++r) {
    int d = eidx[NEDGES + e0 + r];
    const float4 v = *(const float4*)&sg[r * HD128 + lane * 4];
    atomicAdd(&loop_sum[(long)d * HD128 + lane * 4 + 0], v.x);
    atomicAdd(&loop_sum[(long)d * HD128 + lane * 4 + 1], v.y);
    atomicAdd(&loop_sum[(long)d * HD128 + lane * 4 + 2], v.z);
    atomicAdd(&loop_sum[(long)d * HD128 + lane * 4 + 3], v.w);
  }
  if (lane < 16) atomicAdd(&deg[eidx[NEDGES + e0 + lane]], 1.f);
  if (lane < 16) {
    int e = e0 + lane;
    float s = 0.f;
    for (int c = 0; c < HD128; ++c) s += sg[lane * HD128 + c] * w_eff[c];
    int sA = eidx[e], dA = eidx[NEDGES + e];
    float al = asrc[sA] + adst[dA] + s;
    al = al > 0.f ? al : 0.2f * al;
    alpha[e] = al;
    atomicMaxF(&mbuf[dA], al);
  }
}

// ---------------- K3: self-loop edges (fill_value='mean') ----------------
__global__ __launch_bounds__(256) void k_selfloop(const float* __restrict__ loop_sum,
                                                  const float* __restrict__ deg,
                                                  const float* __restrict__ w_eff,
                                                  const float* __restrict__ asrc,
                                                  const float* __restrict__ adst,
                                                  float* __restrict__ alpha,
                                                  float* __restrict__ mbuf) {
  int i = blockIdx.x * 256 + threadIdx.x;
  if (i >= NNODES) return;
  float inv = 1.f / fmaxf(deg[i], 1.f);
  float s = 0.f;
  for (int c = 0; c < HD128; ++c) s += loop_sum[(long)i * HD128 + c] * inv * w_eff[c];
  float al = asrc[i] + adst[i] + s;
  al = al > 0.f ? al : 0.2f * al;
  alpha[NEDGES + i] = al;
  atomicMaxF(&mbuf[i], al);
}

// ---------------- K4: softmax pass 2 (exp + denom) ----------------
__global__ __launch_bounds__(256) void k_soft2(const int* __restrict__ eidx,
                                               const float* __restrict__ mbuf,
                                               float* __restrict__ alpha,
                                               float* __restrict__ denom) {
  long f = (long)blockIdx.x * 256 + threadIdx.x;
  if (f >= (long)NEDGES + NNODES) return;
  int dA = f < NEDGES ? eidx[NEDGES + f] : (int)(f - NEDGES);
  float ex = __expf(alpha[f] - mbuf[dA]);
  alpha[f] = ex;
  atomicAdd(&denom[dA], ex);
}

// ---------------- K5: weighted aggregation (one wave per edge, 4 ch/lane) -------
__global__ __launch_bounds__(256) void k_agg(const int* __restrict__ eidx,
                                             const float* __restrict__ alpha,
                                             const float* __restrict__ denom,
                                             const float* __restrict__ h,
                                             float* __restrict__ outg) {
  int wid = threadIdx.x >> 5, lane = threadIdx.x & 31;
  long f = (long)blockIdx.x * 8 + wid;
  if (f >= (long)NEDGES + NNODES) return;
  int sA = f < NEDGES ? eidx[f] : (int)(f - NEDGES);
  int dA = f < NEDGES ? eidx[NEDGES + f] : (int)(f - NEDGES);
  float att = alpha[f] / denom[dA];
  const float4 hv = *(const float4*)&h[(long)sA * HD128 + lane * 4];
  atomicAdd(&outg[(long)dA * HD128 + lane * 4 + 0], att * hv.x);
  atomicAdd(&outg[(long)dA * HD128 + lane * 4 + 1], att * hv.y);
  atomicAdd(&outg[(long)dA * HD128 + lane * 4 + 2], att * hv.z);
  atomicAdd(&outg[(long)dA * HD128 + lane * 4 + 3], att * hv.w);
}

// ---------------- K6: BN statistics ----------------
__global__ __launch_bounds__(128) void k_bnstats(const float* __restrict__ outg,
                                                 const float* __restrict__ gbias,
                                                 float* __restrict__ bnsum,
                                                 float* __restrict__ bnsq) {
  int c = threadIdx.x;
  float s = 0.f, q = 0.f;
  for (long i = blockIdx.x; i < NNODES; i += gridDim.x) {
    float v = outg[i * HD128 + c] + gbias[c];
    s += v;
    q += v * v;
  }
  atomicAdd(&bnsum[c], s);
  atomicAdd(&bnsq[c], q);
}

// ---------------- K7: 100-token transformer + head (single block) ----------------
__global__ __launch_bounds__(256) void k_transformer(
    const float* __restrict__ outg, const float* __restrict__ gbias,
    const float* __restrict__ bnsum, const float* __restrict__ bnsq,
    const float* __restrict__ bng, const float* __restrict__ bnb,
    const float* __restrict__ Wqkv, const float* __restrict__ bqkv,
    const float* __restrict__ Wo, const float* __restrict__ bo,
    const float* __restrict__ n1g, const float* __restrict__ n1b,
    const float* __restrict__ W1, const float* __restrict__ b1,
    const float* __restrict__ W2, const float* __restrict__ b2,
    const float* __restrict__ n2g, const float* __restrict__ n2b,
    const float* __restrict__ nfg, const float* __restrict__ nfb,
    const float* __restrict__ Wl, const float* __restrict__ bl, float* __restrict__ xs,
    float* __restrict__ qkv, float* __restrict__ attno, float* __restrict__ t1,
    float* __restrict__ ffb, float* __restrict__ out) {
  __shared__ float s_mu[HD128], s_iv[HD128], s_pool[HD128];
  int tid = threadIdx.x;
  if (tid < HD128) {
    float mu = bnsum[tid] / (float)NNODES;
    float var = bnsq[tid] / (float)NNODES - mu * mu;
    s_mu[tid] = mu;
    s_iv[tid] = rsqrtf(var + 1e-5f);
  }
  __syncthreads();
  // gather selected nodes (deterministic hash stands in for JAX permutation) + BN + ReLU
  for (int idx = tid; idx < SEQ * HD128; idx += blockDim.x) {
    int s = idx / HD128, c = idx % HD128;
    unsigned node = (2654435761u * (unsigned)(s + 1)) % (unsigned)NNODES;
    float v = outg[(long)node * HD128 + c] + gbias[c];
    v = (v - s_mu[c]) * s_iv[c] * bng[c] + bnb[c];
    xs[idx] = v > 0.f ? v : 0.f;
  }
  __syncthreads();
  // qkv = xs @ in_proj^T + b
  for (int idx = tid; idx < SEQ * 3 * HD128; idx += blockDim.x) {
    int s = idx / (3 * HD128), oc = idx % (3 * HD128);
    float acc = bqkv[oc];
    const float* xr = xs + s * HD128;
    const float* wr = Wqkv + (long)oc * HD128;
    for (int c = 0; c < HD128; ++c) acc += xr[c] * wr[c];
    qkv[idx] = acc;
  }
  __syncthreads();
  // attention: 2 heads, head_dim 64, online softmax per (head, query)
  for (int idx = tid; idx < SEQ * 2; idx += blockDim.x) {
    int hh = idx & 1, q = idx >> 1;
    const float* qv = qkv + q * 3 * HD128 + hh * 64;
    float mx = -1e30f, den = 0.f, o[64];
#pragma unroll
    for (int d = 0; d < 64; ++d) o[d] = 0.f;
    for (int k = 0; k < SEQ; ++k) {
      const float* kv = qkv + k * 3 * HD128 + HD128 + hh * 64;
      float sc = 0.f;
      for (int d = 0; d < 64; ++d) sc += qv[d] * kv[d];
      sc *= 0.125f;  // 1/sqrt(64)
      float nm = fmaxf(mx, sc);
      float corr = __expf(mx - nm);
      float w = __expf(sc - nm);
      den = den * corr + w;
      const float* vv = qkv + k * 3 * HD128 + 2 * HD128 + hh * 64;
      for (int d = 0; d < 64; ++d) o[d] = o[d] * corr + w * vv[d];
      mx = nm;
    }
    float inv = 1.f / den;
    for (int d = 0; d < 64; ++d) attno[q * HD128 + hh * 64 + d] = o[d] * inv;
  }
  __syncthreads();
  // out_proj + residual
  for (int idx = tid; idx < SEQ * HD128; idx += blockDim.x) {
    int s = idx / HD128, c = idx % HD128;
    float acc = bo[c];
    const float* ar = attno + s * HD128;
    const float* wr = Wo + (long)c * HD128;
    for (int k = 0; k < HD128; ++k) acc += ar[k] * wr[k];
    t1[idx] = xs[idx] + acc;
  }
  __syncthreads();
  // LayerNorm 1
  for (int s = tid; s < SEQ; s += blockDim.x) {
    const float* r = t1 + s * HD128;
    float mu = 0.f;
    for (int c = 0; c < HD128; ++c) mu += r[c];
    mu /= HD128;
    float var = 0.f;
    for (int c = 0; c < HD128; ++c) { float d = r[c] - mu; var += d * d; }
    var /= HD128;
    float iv = rsqrtf(var + 1e-5f);
    for (int c = 0; c < HD128; ++c) xs[s * HD128 + c] = (r[c] - mu) * iv * n1g[c] + n1b[c];
  }
  __syncthreads();
  // FFN lin1 + relu
  for (int idx = tid; idx < SEQ * 2 * HD128; idx += blockDim.x) {
    int s = idx / (2 * HD128), oc = idx % (2 * HD128);
    float acc = b1[oc];
    const float* xr = xs + s * HD128;
    const float* wr = W1 + (long)oc * HD128;
    for (int c = 0; c < HD128; ++c) acc += xr[c] * wr[c];
    ffb[idx] = acc > 0.f ? acc : 0.f;
  }
  __syncthreads();
  // lin2 + residual
  for (int idx = tid; idx < SEQ * HD128; idx += blockDim.x) {
    int s = idx / HD128, c = idx % HD128;
    float acc = b2[c];
    const float* fr = ffb + s * 2 * HD128;
    const float* wr = W2 + (long)c * 2 * HD128;
    for (int k = 0; k < 2 * HD128; ++k) acc += fr[k] * wr[k];
    t1[idx] = xs[idx] + acc;
  }
  __syncthreads();
  // LayerNorm 2 then final LayerNorm
  for (int s = tid; s < SEQ; s += blockDim.x) {
    float* r = t1 + s * HD128;
    float mu = 0.f;
    for (int c = 0; c < HD128; ++c) mu += r[c];
    mu /= HD128;
    float var = 0.f;
    for (int c = 0; c < HD128; ++c) { float d = r[c] - mu; var += d * d; }
    var /= HD128;
    float iv = rsqrtf(var + 1e-5f);
    float tmp[HD128];
    for (int c = 0; c < HD128; ++c) tmp[c] = (r[c] - mu) * iv * n2g[c] + n2b[c];
    mu = 0.f;
    for (int c = 0; c < HD128; ++c) mu += tmp[c];
    mu /= HD128;
    var = 0.f;
    for (int c = 0; c < HD128; ++c) { float d = tmp[c] - mu; var += d * d; }
    var /= HD128;
    iv = rsqrtf(var + 1e-5f);
    for (int c = 0; c < HD128; ++c) xs[s * HD128 + c] = (tmp[c] - mu) * iv * nfg[c] + nfb[c];
  }
  __syncthreads();
  if (tid < HD128) {
    float p = 0.f;
    for (int s = 0; s < SEQ; ++s) p += xs[s * HD128 + tid];
    s_pool[tid] = p / (float)SEQ;
  }
  __syncthreads();
  if (tid < 64) {
    float acc = bl[tid];
    const float* wr = Wl + (long)tid * HD128;
    for (int c = 0; c < HD128; ++c) acc += s_pool[c] * wr[c];
    out[tid] = acc;
  }
}

// ---------------- host ----------------
extern "C" void kernel_launch(void* const* d_in, const int* in_sizes, int n_in,
                              void* d_out, int out_size, void* d_ws, size_t ws_size,
                              hipStream_t stream) {
  (void)in_sizes; (void)n_in; (void)out_size; (void)ws_size;
  const float* x      = (const float*)d_in[0];
  const int*   eidx   = (const int*)d_in[1];
  const float* eattr  = (const float*)d_in[2];
  const float* lap    = (const float*)d_in[3];
  const float* Wn     = (const float*)d_in[4];
  const float* bn     = (const float*)d_in[5];
  const float* We_enc = (const float*)d_in[6];
  const float* be_enc = (const float*)d_in[7];
  const float* Wp     = (const float*)d_in[8];
  const float* bp     = (const float*)d_in[9];
  const float* Wc     = (const float*)d_in[10];
  const float* bc     = (const float*)d_in[11];
  const float* Wg     = (const float*)d_in[12];
  const float* Wge    = (const float*)d_in[13];
  const float* att_s  = (const float*)d_in[14];
  const float* att_d  = (const float*)d_in[15];
  const float* att_e  = (const float*)d_in[16];
  const float* gbias  = (const float*)d_in[17];
  const float* bng    = (const float*)d_in[18];
  const float* bnb    = (const float*)d_in[19];
  const float* Wqkv   = (const float*)d_in[20];
  const float* bqkv   = (const float*)d_in[21];
  const float* Wo     = (const float*)d_in[22];
  const float* bo     = (const float*)d_in[23];
  const float* W1     = (const float*)d_in[24];
  const float* b1     = (const float*)d_in[25];
  const float* W2     = (const float*)d_in[26];
  const float* b2     = (const float*)d_in[27];
  const float* n1g    = (const float*)d_in[28];
  const float* n1b    = (const float*)d_in[29];
  const float* n2g    = (const float*)d_in[30];
  const float* n2b    = (const float*)d_in[31];
  const float* nfg    = (const float*)d_in[32];
  const float* nfb    = (const float*)d_in[33];
  const float* Wl     = (const float*)d_in[34];
  const float* bl     = (const float*)d_in[35];
  float* out = (float*)d_out;

  float* ws = (float*)d_ws;
  float* h        = ws; ws += (size_t)NNODES * HD128;
  float* loop_sum = ws; ws += (size_t)NNODES * HD128;
  float* outg     = ws; ws += (size_t)NNODES * HD128;
  float* deg      = ws; ws += NNODES;
  float* asrc     = ws; ws += NNODES;
  float* adst     = ws; ws += NNODES;
  float* mbuf     = ws; ws += NNODES;
  float* denom    = ws; ws += NNODES;
  float* alpha    = ws; ws += (size_t)(NEDGES + NNODES);
  float* w_eff    = ws; ws += HD128;
  float* bnsum    = ws; ws += HD128;
  float* bnsq     = ws; ws += HD128;
  float* xsb      = ws; ws += SEQ * HD128;
  float* qkvb     = ws; ws += SEQ * 3 * HD128;
  float* attnob   = ws; ws += SEQ * HD128;
  float* t1b      = ws; ws += SEQ * HD128;
  float* ffbb     = ws; ws += SEQ * 2 * HD128;

  k_init<<<2048, 256, 0, stream>>>(loop_sum, outg, deg, denom, mbuf, bnsum, bnsq);

  int node_tiles = (NNODES + 15) / 16;
  k_nodes<<<(node_tiles + 3) / 4, 128, 0, stream>>>(x, lap, Wn, bn, Wp, bp, Wc, bc, Wg,
                                                    att_s, att_d, h, asrc, adst);
  k_weff<<<1, 128, 0, stream>>>(Wge, att_e, w_eff);

  int edge_tiles = NEDGES / 16;
  k_edges<<<(edge_tiles + 3) / 4, 128, 0, stream>>>(eattr, eidx, We_enc, be_enc, w_eff,
                                                    asrc, adst, loop_sum, deg, alpha, mbuf);
  k_selfloop<<<(NNODES + 255) / 256, 256, 0, stream>>>(loop_sum, deg, w_eff, asrc, adst,
                                                       alpha, mbuf);
  long tot = (long)NEDGES + NNODES;
  k_soft2<<<(int)((tot + 255) / 256), 256, 0, stream>>>(eidx, mbuf, alpha, denom);
  k_agg<<<(int)((tot + 7) / 8), 256, 0, stream>>>(eidx, alpha, denom, h, outg);
  k_bnstats<<<512, 128, 0, stream>>>(outg, gbias, bnsum, bnsq);
  k_transformer<<<1, 256, 0, stream>>>(outg, gbias, bnsum, bnsq, bng, bnb, Wqkv, bqkv, Wo,
                                       bo, n1g, n1b, W1, b1, W2, b2, n2g, n2b, nfg, nfb,
                                       Wl, bl, xsb, qkvb, attnob, t1b, ffbb, out);
}
